// TGAT_15161234555327
// MI455X (gfx1250) — matrix operand order repaired and checked by reading, not compiled, not run on hardware
//
#include <hip/hip_runtime.h>
#include <hip/hip_bf16.h>

typedef __attribute__((ext_vector_type(16))) __bf16 v16bf;
typedef __attribute__((ext_vector_type(8)))  float  v8f;

#define LEAKY_F 0.2f
#define LN_EPS_F 1e-5f

__device__ __forceinline__ v8f wmma_bf16(v16bf a, v16bf b, v8f c) {
    return __builtin_amdgcn_wmma_f32_16x16x32_bf16(false, a, false, b,
                                                   (short)0, c, false, false);
}

// Packed-B layout: tile (nt,kc) -> 32 lanes x 16 bf16 contiguous.
// packed[((nt*nK+kc)*32+lane)*16 + e] = W[nt*16 + lane%16][kc*32 + 8*(lane/16) + e + (e>=8 ? 8 : 0)]
__device__ __forceinline__ v16bf load_packedB(const __bf16* __restrict__ pw,
                                              int nt, int kc, int nK, int lane) {
    const __bf16* p = pw + ((((size_t)nt * nK + kc) * 32 + lane) << 4);
    return *(const v16bf*)p;   // 32B -> two global_load_b128
}

// A tile element mapping (16x32 bf16): VGPR pair r, lane half lh:
// K offset = (r<4 ? 2r+8*lh : 16+2*(r-4)+8*lh), elements {K, K+1}
__device__ __forceinline__ int a_pair_k(int r, int lh) {
    return (r < 4) ? (2 * r + 8 * lh) : (16 + 2 * (r - 4) + 8 * lh);
}

// kin row = [ src(DN) | edge(DE) | te(DT) | pad ]  (boundaries all even)
template<int DN, int DE, int DT>
__device__ __forceinline__ v16bf build_kin_tile(const float* __restrict__ srow,
                                                const float* __restrict__ erow,
                                                float dtv,
                                                const float* __restrict__ tw,
                                                const float* __restrict__ tb,
                                                int kbase, int lh) {
    v16bf a;
#pragma unroll
    for (int r = 0; r < 8; ++r) {
        int c = kbase + a_pair_k(r, lh);
        float x0, x1;
        if (c < DN)                 { x0 = srow[c];        x1 = srow[c + 1]; }
        else if (c < DN + DE)       { x0 = erow[c - DN];   x1 = erow[c - DN + 1]; }
        else if (c < DN + DE + DT)  { int j = c - DN - DE;
                                      x0 = __cosf(dtv * tw[j]     + tb[j]);
                                      x1 = __cosf(dtv * tw[j + 1] + tb[j + 1]); }
        else                        { x0 = 0.f; x1 = 0.f; }
        a[2 * r]     = (__bf16)x0;
        a[2 * r + 1] = (__bf16)x1;
    }
    return a;
}

// q row = [ dst(DN) | cos(time_b)(DT) | pad ]
template<int DN, int DT>
__device__ __forceinline__ v16bf build_q_tile(const float* __restrict__ drow,
                                              const float* __restrict__ tb,
                                              int kbase, int lh) {
    v16bf a;
#pragma unroll
    for (int r = 0; r < 8; ++r) {
        int c = kbase + a_pair_k(r, lh);
        float x0, x1;
        if (c < DN)            { x0 = drow[c]; x1 = drow[c + 1]; }
        else if (c < DN + DT)  { int j = c - DN;
                                 x0 = __cosf(tb[j]); x1 = __cosf(tb[j + 1]); }
        else                   { x0 = 0.f; x1 = 0.f; }
        a[2 * r]     = (__bf16)x0;
        a[2 * r + 1] = (__bf16)x1;
    }
    return a;
}

// o row = [ out(128, from LDS f32) | dst(DN) | pad ]
template<int DN>
__device__ __forceinline__ v16bf build_o_tile(const float* __restrict__ orow,
                                              const float* __restrict__ drow,
                                              int kbase, int lh) {
    v16bf a;
#pragma unroll
    for (int r = 0; r < 8; ++r) {
        int c = kbase + a_pair_k(r, lh);
        float x0, x1;
        if (c < 128)            { x0 = orow[c];        x1 = orow[c + 1]; }
        else if (c < 128 + DN)  { x0 = drow[c - 128];  x1 = drow[c - 127]; }
        else                    { x0 = 0.f; x1 = 0.f; }
        a[2 * r]     = (__bf16)x0;
        a[2 * r + 1] = (__bf16)x1;
    }
    return a;
}

// Fused TGAT attention layer: 16 dst nodes per block, 256 threads (8 wave32).
// DO = 128, H = 2 heads, K_ = 10 neighbors.
template<int DN, int DE, int DT>
__global__ void __launch_bounds__(256)
attn_layer_kernel(const float* __restrict__ dst_h,   // (N, DN)
                  const float* __restrict__ src_h,   // (N*10, DN)
                  const float* __restrict__ edge_f,  // (N*10, DE)
                  const float* __restrict__ dt,      // (N*10)
                  const float* __restrict__ time_w, const float* __restrict__ time_b,
                  const float* __restrict__ bq, const float* __restrict__ bk,
                  const float* __restrict__ bv, const float* __restrict__ bo,
                  const float* __restrict__ ln_g, const float* __restrict__ ln_b,
                  const __bf16* __restrict__ pWq, const __bf16* __restrict__ pWk,
                  const __bf16* __restrict__ pWv, const __bf16* __restrict__ pWo,
                  float* __restrict__ h_out,         // (N, 128)
                  int N) {
    constexpr int K_ = 10;
    constexpr int DKIN = DN + DE + DT;
    constexpr int NKC_K = (DKIN + 31) / 32;
    constexpr int NKC_Q = (DN + DT + 31) / 32;
    constexpr int NKC_O = (128 + DN + 31) / 32;

    extern __shared__ float smem[];
    float* lds_q   = smem;                   // 16*128
    float* lds_k   = lds_q   + 16 * 128;     // 160*128
    float* lds_v   = lds_k   + 160 * 128;    // 160*128
    float* lds_att = lds_v   + 160 * 128;    // 160*2
    float* lds_out = lds_att + 160 * 2;      // 16*128
    float* lds_st  = lds_out + 16 * 128;     // 16*2

    const int tid  = threadIdx.x;
    const int lane = tid & 31;
    const int wave = tid >> 5;
    const int lm   = lane & 15;
    const int lh   = lane >> 4;

    const int node0 = blockIdx.x * 16;
    const int row0  = node0 * K_;
    const int NR    = N * K_;

    // ---- Part A: q = [dst, zero_te] @ Wq^T  (wave w -> N-tile w)
    {
        int nodeG = min(node0 + lm, N - 1);
        const float* drow = dst_h + (size_t)nodeG * DN;
        v8f acc = {0.f, 0.f, 0.f, 0.f, 0.f, 0.f, 0.f, 0.f};
        for (int kc = 0; kc < NKC_Q; ++kc) {
            v16bf a = build_q_tile<DN, DT>(drow, time_b, kc * 32, lh);
            v16bf b = load_packedB(pWq, wave, kc, NKC_Q, lane);
            acc = wmma_bf16(a, b, acc);
        }
        float bias = bq[wave * 16 + lm];
#pragma unroll
        for (int r = 0; r < 8; ++r)
            lds_q[(r + 8 * lh) * 128 + wave * 16 + lm] = acc[r] + bias;
    }

    // ---- Part B: k,v for 160 rows (10 M-tiles, strided over 8 waves)
    for (int mt = wave; mt < 10; mt += 8) {
        int rowG = min(row0 + mt * 16 + lm, NR - 1);
        const float* srow = src_h  + (size_t)rowG * DN;
        const float* erow = edge_f + (size_t)rowG * DE;
        float dtv = dt[rowG];
        v8f accK[8], accV[8];
        v8f z = {0.f, 0.f, 0.f, 0.f, 0.f, 0.f, 0.f, 0.f};
#pragma unroll
        for (int nt = 0; nt < 8; ++nt) { accK[nt] = z; accV[nt] = z; }
        for (int kc = 0; kc < NKC_K; ++kc) {
            v16bf a = build_kin_tile<DN, DE, DT>(srow, erow, dtv, time_w, time_b,
                                                 kc * 32, lh);
#pragma unroll
            for (int nt = 0; nt < 8; ++nt) {
                v16bf bkt = load_packedB(pWk, nt, kc, NKC_K, lane);
                accK[nt] = wmma_bf16(a, bkt, accK[nt]);
                v16bf bvt = load_packedB(pWv, nt, kc, NKC_K, lane);
                accV[nt] = wmma_bf16(a, bvt, accV[nt]);
            }
        }
#pragma unroll
        for (int nt = 0; nt < 8; ++nt) {
            float biask = bk[nt * 16 + lm];
            float biasv = bv[nt * 16 + lm];
#pragma unroll
            for (int r = 0; r < 8; ++r) {
                int rr = mt * 16 + r + 8 * lh;
                lds_k[rr * 128 + nt * 16 + lm] = accK[nt][r] + biask;
                lds_v[rr * 128 + nt * 16 + lm] = accV[nt][r] + biasv;
            }
        }
    }
    __syncthreads();

    // ---- Part C: per-(row,head) scores + leaky relu, then softmax over 10
    for (int idx = tid; idx < 160 * 2; idx += 256) {
        int row = idx >> 1, head = idx & 1;
        int nd = row / K_;
        const float* qp = lds_q + nd * 128 + head * 64;
        const float* kp = lds_k + row * 128 + head * 64;
        float s = 0.f;
#pragma unroll 8
        for (int d = 0; d < 64; ++d) s += qp[d] * kp[d];
        lds_att[idx] = (s > 0.f) ? s : LEAKY_F * s;
    }
    __syncthreads();
    if (tid < 32) {
        int nd = tid >> 1, head = tid & 1;
        float m = -1e30f;
        for (int kn = 0; kn < K_; ++kn)
            m = fmaxf(m, lds_att[(nd * K_ + kn) * 2 + head]);
        float e[K_]; float sum = 0.f;
        for (int kn = 0; kn < K_; ++kn) {
            e[kn] = __expf(lds_att[(nd * K_ + kn) * 2 + head] - m);
            sum += e[kn];
        }
        float inv = 1.f / sum;
        for (int kn = 0; kn < K_; ++kn)
            lds_att[(nd * K_ + kn) * 2 + head] = e[kn] * inv;
    }
    __syncthreads();

    // ---- Part D: out = att @ v   (16 x 128)
    for (int idx = tid; idx < 16 * 128; idx += 256) {
        int nd = idx >> 7, c = idx & 127, head = c >> 6;
        float o = 0.f;
#pragma unroll
        for (int kn = 0; kn < K_; ++kn)
            o += lds_att[(nd * K_ + kn) * 2 + head] *
                 lds_v[(nd * K_ + kn) * 128 + c];
        lds_out[idx] = o;
    }
    __syncthreads();

    // ---- Part E: rst = relu([out, dst] @ Wo^T)  (wave w -> N-tile w)
    v8f accO = {0.f, 0.f, 0.f, 0.f, 0.f, 0.f, 0.f, 0.f};
    {
        int nodeG = min(node0 + lm, N - 1);
        const float* drow = dst_h + (size_t)nodeG * DN;
        const float* orow = lds_out + lm * 128;
        for (int kc = 0; kc < NKC_O; ++kc) {
            v16bf a = build_o_tile<DN>(orow, drow, kc * 32, lh);
            v16bf b = load_packedB(pWo, wave, kc, NKC_O, lane);
            accO = wmma_bf16(a, b, accO);
        }
    }
    __syncthreads();   // everyone finished reading lds_out
    {
        float bias = bo[wave * 16 + lm];
#pragma unroll
        for (int r = 0; r < 8; ++r)
            lds_out[(r + 8 * lh) * 128 + wave * 16 + lm] =
                fmaxf(accO[r] + bias, 0.f);
    }
    __syncthreads();

    // ---- Part F: LayerNorm + store
    if (tid < 16) {
        float mu = 0.f;
        for (int c = 0; c < 128; ++c) mu += lds_out[tid * 128 + c];
        mu *= (1.f / 128.f);
        float var = 0.f;
        for (int c = 0; c < 128; ++c) {
            float d = lds_out[tid * 128 + c] - mu; var += d * d;
        }
        var *= (1.f / 128.f);
        lds_st[tid * 2]     = mu;
        lds_st[tid * 2 + 1] = rsqrtf(var + LN_EPS_F);
    }
    __syncthreads();
    for (int idx = tid; idx < 16 * 128; idx += 256) {
        int r = idx >> 7, c = idx & 127;
        int nd = node0 + r;
        if (nd < N) {
            float mu = lds_st[r * 2], rs = lds_st[r * 2 + 1];
            h_out[(size_t)nd * 128 + c] =
                (lds_out[idx] - mu) * rs * ln_g[c] + ln_b[c];
        }
    }
}

// Pack W (128 x Din, row-major) -> WMMA-B bf16 tile layout (zero-padded K).
__global__ void pack_weight_kernel(const float* __restrict__ W,
                                   __bf16* __restrict__ out,
                                   int nK, int Din) {
    int idx = blockIdx.x * blockDim.x + threadIdx.x;
    int total = 8 * nK * 512;
    if (idx >= total) return;
    int e    = idx & 15;
    int lane = (idx >> 4) & 31;
    int t    = idx >> 9;
    int kc   = t % nK;
    int nt   = t / nK;
    int n = nt * 16 + (lane & 15);
    int k = kc * 32 + ((lane >> 4) << 3) + e + ((e >> 3) << 3);
    float v = (k < Din) ? W[(size_t)n * Din + k] : 0.f;
    out[idx] = (__bf16)v;
}

// score = Wo @ relu(a@Ws^T + bs + b@Wd^T + bd) + bo ; one block per row.
__global__ void __launch_bounds__(128)
edge_pred_kernel(const float* __restrict__ A, const float* __restrict__ Bm,
                 const float* __restrict__ Ws, const float* __restrict__ bs,
                 const float* __restrict__ Wd, const float* __restrict__ bd,
                 const float* __restrict__ Wo, const float* __restrict__ bo,
                 float* __restrict__ out) {
    __shared__ float red[128];
    int row = blockIdx.x, c = threadIdx.x;
    const float* ar = A  + (size_t)row * 128;
    const float* br = Bm + (size_t)row * 128;
    const float* ws = Ws + (size_t)c * 128;
    const float* wd = Wd + (size_t)c * 128;
    float h = bs[c] + bd[c];
#pragma unroll 8
    for (int j = 0; j < 128; ++j) h += ar[j] * ws[j] + br[j] * wd[j];
    red[c] = fmaxf(h, 0.f) * Wo[c];
    __syncthreads();
    for (int s = 64; s > 0; s >>= 1) {
        if (c < s) red[c] += red[c + s];
        __syncthreads();
    }
    if (c == 0) out[row] = red[0] + bo[0];
}

extern "C" void kernel_launch(void* const* d_in, const int* in_sizes, int n_in,
                              void* d_out, int out_size, void* d_ws, size_t ws_size,
                              hipStream_t stream) {
    const int Bn = 1000, K_ = 10;
    const int R  = 3 * Bn;            // 3000
    const int N1 = R * (1 + K_);      // 33000

    // setup_inputs() insertion order, nested dicts in insertion order
    const float* nf_dst = (const float*)d_in[0];
    const float* nf_src = (const float*)d_in[1];
    const float* ef1    = (const float*)d_in[2];
    const float* dt1    = (const float*)d_in[3];
    const float* ef2    = (const float*)d_in[4];
    const float* dt2    = (const float*)d_in[5];
    // params0: time_w,time_b,Wq,bq,Wk,bk,Wv,bv,Wo,bo,ln_g,ln_b
    const float* p0_tw = (const float*)d_in[6];
    const float* p0_tb = (const float*)d_in[7];
    const float* p0_Wq = (const float*)d_in[8];
    const float* p0_bq = (const float*)d_in[9];
    const float* p0_Wk = (const float*)d_in[10];
    const float* p0_bk = (const float*)d_in[11];
    const float* p0_Wv = (const float*)d_in[12];
    const float* p0_bv = (const float*)d_in[13];
    const float* p0_Wo = (const float*)d_in[14];
    const float* p0_bo = (const float*)d_in[15];
    const float* p0_g  = (const float*)d_in[16];
    const float* p0_b  = (const float*)d_in[17];
    const float* p1_tw = (const float*)d_in[18];
    const float* p1_tb = (const float*)d_in[19];
    const float* p1_Wq = (const float*)d_in[20];
    const float* p1_bq = (const float*)d_in[21];
    const float* p1_Wk = (const float*)d_in[22];
    const float* p1_bk = (const float*)d_in[23];
    const float* p1_Wv = (const float*)d_in[24];
    const float* p1_bv = (const float*)d_in[25];
    const float* p1_Wo = (const float*)d_in[26];
    const float* p1_bo = (const float*)d_in[27];
    const float* p1_g  = (const float*)d_in[28];
    const float* p1_b  = (const float*)d_in[29];
    const float* pr_Ws = (const float*)d_in[30];
    const float* pr_bs = (const float*)d_in[31];
    const float* pr_Wd = (const float*)d_in[32];
    const float* pr_bd = (const float*)d_in[33];
    const float* pr_Wo = (const float*)d_in[34];
    const float* pr_bo = (const float*)d_in[35];

    // ---- workspace carve (all 32B aligned)
    char* ws = (char*)d_ws;
    float* h1 = (float*)ws;  ws += (size_t)N1 * 128 * sizeof(float);
    float* h2 = (float*)ws;  ws += (size_t)R  * 128 * sizeof(float);
    const int NKQ0 = 10, NKK0 = 15, NKO0 = 10;   // layer1: 300, 472, 300 cols
    const int NKQ1 = 8,  NKK1 = 14, NKO1 = 8;    // layer2: 256, 428, 256 cols
    __bf16* pWq0 = (__bf16*)ws; ws += (size_t)8 * NKQ0 * 512 * 2;
    __bf16* pWk0 = (__bf16*)ws; ws += (size_t)8 * NKK0 * 512 * 2;
    __bf16* pWv0 = (__bf16*)ws; ws += (size_t)8 * NKK0 * 512 * 2;
    __bf16* pWo0 = (__bf16*)ws; ws += (size_t)8 * NKO0 * 512 * 2;
    __bf16* pWq1 = (__bf16*)ws; ws += (size_t)8 * NKQ1 * 512 * 2;
    __bf16* pWk1 = (__bf16*)ws; ws += (size_t)8 * NKK1 * 512 * 2;
    __bf16* pWv1 = (__bf16*)ws; ws += (size_t)8 * NKK1 * 512 * 2;
    __bf16* pWo1 = (__bf16*)ws; ws += (size_t)8 * NKO1 * 512 * 2;

    // ---- weight packing (bf16, WMMA-B layout)
    auto pack = [&](const float* W, __bf16* P, int nK, int Din) {
        int total = 8 * nK * 512;
        pack_weight_kernel<<<(total + 255) / 256, 256, 0, stream>>>(W, P, nK, Din);
    };
    pack(p0_Wq, pWq0, NKQ0, 300);
    pack(p0_Wk, pWk0, NKK0, 472);
    pack(p0_Wv, pWv0, NKK0, 472);
    pack(p0_Wo, pWo0, NKO0, 300);
    pack(p1_Wq, pWq1, NKQ1, 256);
    pack(p1_Wk, pWk1, NKK1, 428);
    pack(p1_Wv, pWv1, NKK1, 428);
    pack(p1_Wo, pWo1, NKO1, 256);

    // ---- fused attention layers
    const int SMEM = (16 * 128 + 160 * 128 + 160 * 128 + 160 * 2 + 16 * 128 + 32)
                     * (int)sizeof(float);   // ~178 KB of the 320 KB WGP LDS
    (void)hipFuncSetAttribute(
        reinterpret_cast<const void*>(&attn_layer_kernel<172, 172, 128>),
        hipFuncAttributeMaxDynamicSharedMemorySize, SMEM);
    (void)hipFuncSetAttribute(
        reinterpret_cast<const void*>(&attn_layer_kernel<128, 172, 128>),
        hipFuncAttributeMaxDynamicSharedMemorySize, SMEM);

    attn_layer_kernel<172, 172, 128><<<(N1 + 15) / 16, 256, SMEM, stream>>>(
        nf_dst, nf_src, ef1, dt1, p0_tw, p0_tb,
        p0_bq, p0_bk, p0_bv, p0_bo, p0_g, p0_b,
        pWq0, pWk0, pWv0, pWo0, h1, N1);

    attn_layer_kernel<128, 172, 128><<<(R + 15) / 16, 256, SMEM, stream>>>(
        h1, h1 + (size_t)R * 128, ef2, dt2, p1_tw, p1_tb,
        p1_bq, p1_bk, p1_bv, p1_bo, p1_g, p1_b,
        pWq1, pWk1, pWv1, pWo1, h2, R);

    // ---- edge predictor: pos = (src,dst), neg = (src,neg)
    float* outp = (float*)d_out;
    edge_pred_kernel<<<Bn, 128, 0, stream>>>(
        h2, h2 + (size_t)Bn * 128,
        pr_Ws, pr_bs, pr_Wd, pr_bd, pr_Wo, pr_bo, outp);
    edge_pred_kernel<<<Bn, 128, 0, stream>>>(
        h2, h2 + (size_t)2 * Bn * 128,
        pr_Ws, pr_bs, pr_Wd, pr_bd, pr_Wo, pr_bo, outp + Bn);
}
